// GCNNet2_38551626449341
// MI455X (gfx1250) — matrix operand order repaired and checked by reading, not compiled, not run on hardware
//
#include <hip/hip_runtime.h>

#define D 128
#define EPSV 1e-5f

typedef __attribute__((ext_vector_type(16))) __bf16 v16bf;
typedef __attribute__((ext_vector_type(8)))  float  v8f;
typedef int v4i __attribute__((vector_size(4 * sizeof(int))));

#if defined(__HIP_DEVICE_COMPILE__) && __has_builtin(__builtin_amdgcn_global_load_async_to_lds_b128)
#define USE_ASYNC_LDS 1
#else
#define USE_ASYNC_LDS 0
#endif

__device__ __forceinline__ __bf16 f2bf(float f) {
  unsigned u = __builtin_bit_cast(unsigned, f);
  unsigned short s = (unsigned short)((u + 0x7FFFu + ((u >> 16) & 1u)) >> 16);
  return __builtin_bit_cast(__bf16, s);
}

// Transpose+convert weights to bf16: Wt[m][n*D+k] = W_m[k][n]; m=0 emb, m=1..3 gcn layers 0..2
__global__ void k_prep_weights(const float* __restrict__ emb_w,
                               const float* __restrict__ gcn_w,
                               __bf16* __restrict__ wt) {
  int idx = blockIdx.x * blockDim.x + threadIdx.x;
  if (idx >= 4 * D * D) return;
  int m = idx >> 14;
  int r = idx & (D * D - 1);
  int n = r >> 7;
  int k = r & (D - 1);
  const float* src = (m == 0) ? emb_w : (gcn_w + (size_t)(m - 1) * D * D);
  wt[(size_t)m * D * D + n * D + k] = f2bf(src[k * D + n]);
}

// v_l[d] = sum_c skip_w[l][d][c] * readout_w[l*D+c]   (folds skip GEMMs + readout into GEMV)
__global__ void k_compute_v(const float* __restrict__ skip_w,
                            const float* __restrict__ readout_w,
                            float* __restrict__ v) {
  int d = threadIdx.x;  // 128 threads
  for (int l = 0; l < 4; ++l) {
    const float* w = skip_w + (size_t)l * D * D + (size_t)d * D;
    const float* r = readout_w + l * D;
    float s = 0.f;
    for (int c = 0; c < D; ++c) s += w[c] * r[c];
    v[l * D + d] = s;
  }
}

__global__ void k_deg(const long long* __restrict__ col, float* __restrict__ deg, int E_) {
  int e = blockIdx.x * blockDim.x + threadIdx.x;
  if (e < E_) atomicAdd(&deg[col[e]], 1.0f);
}

__global__ void k_dinv(float* __restrict__ deg, int N_) {
  int i = blockIdx.x * blockDim.x + threadIdx.x;
  if (i < N_) deg[i] = rsqrtf(deg[i] + 1.0f);  // +1 self loop; always > 0
}

// C[N x 128] = A[N x 128] @ W[128 x 128] (+bias). One wave -> 16 rows x 128 cols.
// Weight matrix (32 KB bf16) staged in LDS once per block via async global->LDS copy.
__global__ void k_gemm_bf16(const float* __restrict__ A, const __bf16* __restrict__ Wt,
                            const float* __restrict__ bias, float* __restrict__ C, int ntiles) {
  __shared__ __bf16 wlds[D * D];  // 32 KB of 320 KB WGP LDS

  // Cooperative stage-in: 2048 x 16B chunks, 8 per thread (256 threads)
  {
    const int t = threadIdx.x;
#if USE_ASYNC_LDS
    const char* gsrc = (const char*)Wt;
    char* ldst = (char*)wlds;
#pragma unroll
    for (int j = 0; j < 8; ++j) {
      const int byteoff = (t + j * 256) * 16;
      __builtin_amdgcn_global_load_async_to_lds_b128(
          (__attribute__((address_space(1))) v4i*)(gsrc + byteoff),
          (__attribute__((address_space(3))) v4i*)(ldst + byteoff), 0, 0);
    }
    asm volatile("s_wait_asynccnt 0x0" ::: "memory");
#else
    const float4* gsrc = (const float4*)Wt;
    float4* ldst = (float4*)wlds;
#pragma unroll
    for (int j = 0; j < 8; ++j) ldst[t + j * 256] = gsrc[t + j * 256];
#endif
  }
  __syncthreads();

  int gid  = blockIdx.x * blockDim.x + threadIdx.x;
  int tile = gid >> 5;
  if (tile >= ntiles) return;               // wave-uniform: EXEC all-1s at every WMMA
  int lane = threadIdx.x & 31;
  int hlf  = lane >> 4;
  int lrow = lane & 15;
  const float* arow = A + (size_t)(tile * 16 + lrow) * D;

  v8f acc[8] = {};
  for (int kc = 0; kc < 4; ++kc) {
    // A-frag (16-bit A 16x32 layout): K = hlf*16 + 0..15 contiguous (matches B)
    const float* a0 = arow + kc * 32 + hlf * 16;
    const float* a1 = a0 + 8;
    v16bf af;
#pragma unroll
    for (int j = 0; j < 8; ++j) { af[j] = f2bf(a0[j]); af[8 + j] = f2bf(a1[j]); }
#pragma unroll
    for (int nc = 0; nc < 8; ++nc) {
      // B-frag from LDS: col = lane&15, K = hlf*16 + 0..15 contiguous (2x ds_load_b128)
      const v16bf bf_ = *(const v16bf*)(wlds + (nc * 16 + lrow) * D + kc * 32 + hlf * 16);
      acc[nc] = __builtin_amdgcn_wmma_f32_16x16x32_bf16(
          false, af, false, bf_, (short)0, acc[nc], false, false);
    }
  }
#pragma unroll
  for (int nc = 0; nc < 8; ++nc) {
    int colI = nc * 16 + lrow;
    float b  = bias ? bias[colI] : 0.f;
#pragma unroll
    for (int m = 0; m < 8; ++m) {
      int orow = tile * 16 + m + hlf * 8;   // C layout: VGPR m -> row m (+8 for high half)
      C[(size_t)orow * D + colI] = acc[nc][m] + b;
    }
  }
}

// Wave per edge: agg[col] += hl[row] * dinv[row]*dinv[col]; each lane moves float4
__global__ void k_scatter(const float* __restrict__ hl, const long long* __restrict__ rowi,
                          const long long* __restrict__ coli, const float* __restrict__ dinv,
                          float* __restrict__ agg, int E_) {
  int gid = blockIdx.x * blockDim.x + threadIdx.x;
  int e = gid >> 5;
  if (e >= E_) return;
  int lane = threadIdx.x & 31;
  long long r = rowi[e], c = coli[e];
  float nrm = dinv[r] * dinv[c];
  float4 m = ((const float4*)(hl + (size_t)r * D))[lane];
  float* dst = agg + (size_t)c * D + lane * 4;
  atomicAdd(dst + 0, m.x * nrm);
  atomicAdd(dst + 1, m.y * nrm);
  atomicAdd(dst + 2, m.z * nrm);
  atomicAdd(dst + 3, m.w * nrm);
}

// Self loop: agg[i] += hl[i] * dinv[i]^2  (runs after scatter, stream-ordered, no atomics)
__global__ void k_selfloop(const float* __restrict__ hl, const float* __restrict__ dinv,
                           float* __restrict__ agg, int N_) {
  int gid = blockIdx.x * blockDim.x + threadIdx.x;
  int i = gid >> 5;
  if (i >= N_) return;
  int lane = threadIdx.x & 31;
  float w = dinv[i] * dinv[i];
  float4 m = ((const float4*)(hl + (size_t)i * D))[lane];
  float4* dst = (float4*)(agg + (size_t)i * D) + lane;
  float4 d = *dst;
  d.x += m.x * w; d.y += m.y * w; d.z += m.z * w; d.w += m.w * w;
  *dst = d;
}

// Per-column sum & sumsq partials, atomically merged into sums[0..127], sums[128..255]
__global__ void k_bn_stats(const float* __restrict__ agg, float* __restrict__ sums, int N_) {
  int d  = threadIdx.x;  // 128 threads
  int r0 = blockIdx.x * 512;
  int r1 = r0 + 512; if (r1 > N_) r1 = N_;
  float s = 0.f, s2 = 0.f;
  for (int r = r0; r < r1; ++r) {
    float v = agg[(size_t)r * D + d];
    s += v; s2 += v * v;
  }
  atomicAdd(&sums[d], s);
  atomicAdd(&sums[D + d], s2);
}

__global__ void k_bn_apply(const float* __restrict__ agg, const float* __restrict__ sums,
                           const float* __restrict__ g, const float* __restrict__ b,
                           float* __restrict__ out, int N_) {
  int idx = blockIdx.x * blockDim.x + threadIdx.x;
  if (idx >= N_ * D) return;
  int d = idx & (D - 1);
  float inv  = 1.0f / (float)N_;
  float mean = sums[d] * inv;
  float var  = sums[D + d] * inv - mean * mean;
  float y = (agg[idx] - mean) * rsqrtf(var + EPSV) * g[d] + b[d];
  out[idx] = fmaxf(y, 0.f);
}

// logits[i] = sum_l feats[l][i] . v_l ; out = [logits | sigmoid(logits)]
__global__ void k_readout(const float* __restrict__ feats, const float* __restrict__ v,
                          float* __restrict__ out, int N_) {
  int gid = blockIdx.x * blockDim.x + threadIdx.x;
  int i = gid >> 5;
  if (i >= N_) return;
  int lane = threadIdx.x & 31;
  float s = 0.f;
#pragma unroll
  for (int l = 0; l < 4; ++l) {
    float4 f = ((const float4*)(feats + ((size_t)l * N_ + i) * D))[lane];
    float4 w = ((const float4*)(v + l * D))[lane];
    s += f.x * w.x + f.y * w.y + f.z * w.z + f.w * w.w;
  }
#pragma unroll
  for (int off = 16; off > 0; off >>= 1) s += __shfl_down(s, off, 32);
  if (lane == 0) { out[i] = s; out[N_ + i] = 1.0f / (1.0f + __expf(-s)); }
}

extern "C" void kernel_launch(void* const* d_in, const int* in_sizes, int n_in,
                              void* d_out, int out_size, void* d_ws, size_t ws_size,
                              hipStream_t stream) {
  const float* x         = (const float*)d_in[0];
  const long long* ei    = (const long long*)d_in[1];   // int64 (2,E)
  const float* emb_w     = (const float*)d_in[2];
  const float* emb_b     = (const float*)d_in[3];
  const float* gcn_w     = (const float*)d_in[4];
  // d_in[5] (gcn_b) cancels under BatchNorm mean-subtraction -> unused
  const float* bn_g      = (const float*)d_in[6];
  const float* bn_b      = (const float*)d_in[7];
  const float* skip_w    = (const float*)d_in[8];
  const float* readout_w = (const float*)d_in[9];
  float* out = (float*)d_out;

  const int N_ = in_sizes[0] / D;
  const int E_ = in_sizes[1] / 2;
  const long long* rowi = ei;
  const long long* coli = ei + E_;

  char* ws = (char*)d_ws;
  size_t off = 0;
  auto alloc = [&](size_t bytes) -> char* {
    char* p = ws + off; off += (bytes + 255) & ~(size_t)255; return p;
  };
  float*  feats = (float*)alloc((size_t)4 * N_ * D * sizeof(float));  // feats[0..3]
  float*  hl    = (float*)alloc((size_t)N_ * D * sizeof(float));
  float*  agg   = (float*)alloc((size_t)N_ * D * sizeof(float));
  float*  dinv  = (float*)alloc((size_t)N_ * sizeof(float));
  float*  sums  = (float*)alloc(2 * D * sizeof(float));
  float*  vvec  = (float*)alloc(4 * D * sizeof(float));
  __bf16* wt    = (__bf16*)alloc((size_t)4 * D * D * sizeof(__bf16));
  (void)ws_size; (void)n_in; (void)out_size;

  // Weight prep (tiny)
  k_prep_weights<<<(4 * D * D + 255) / 256, 256, 0, stream>>>(emb_w, gcn_w, wt);
  k_compute_v<<<1, D, 0, stream>>>(skip_w, readout_w, vvec);

  // Degrees -> dinv
  (void)hipMemsetAsync(dinv, 0, (size_t)N_ * sizeof(float), stream);
  k_deg<<<(E_ + 255) / 256, 256, 0, stream>>>(coli, dinv, E_);
  k_dinv<<<(N_ + 255) / 256, 256, 0, stream>>>(dinv, N_);

  // Embedding: feats[0] = x @ emb_w + emb_b
  const int ntiles = (N_ + 15) / 16;
  const int gemm_blocks = (ntiles * 32 + 255) / 256;
  k_gemm_bf16<<<gemm_blocks, 256, 0, stream>>>(x, wt, emb_b, feats, ntiles);

  const int wblocksE = (E_ * 32 + 255) / 256;
  const int wblocksN = (N_ * 32 + 255) / 256;
  for (int l = 0; l < 3; ++l) {  // layer 3 output unused by skip -> skipped entirely
    const float* fin = feats + (size_t)l * N_ * D;
    k_gemm_bf16<<<gemm_blocks, 256, 0, stream>>>(fin, wt + (size_t)(l + 1) * D * D,
                                                 (const float*)nullptr, hl, ntiles);
    (void)hipMemsetAsync(agg, 0, (size_t)N_ * D * sizeof(float), stream);
    k_scatter<<<wblocksE, 256, 0, stream>>>(hl, rowi, coli, dinv, agg, E_);
    k_selfloop<<<wblocksN, 256, 0, stream>>>(hl, dinv, agg, N_);
    (void)hipMemsetAsync(sums, 0, 2 * D * sizeof(float), stream);
    k_bn_stats<<<(N_ + 511) / 512, D, 0, stream>>>(agg, sums, N_);
    k_bn_apply<<<(N_ * D + 255) / 256, 256, 0, stream>>>(
        agg, sums, bn_g + l * D, bn_b + l * D, feats + (size_t)(l + 1) * N_ * D, N_);
  }

  k_readout<<<wblocksN, 256, 0, stream>>>(feats, vvec, out, N_);
}
